// rec_16484084483545
// MI455X (gfx1250) — compile-verified, matrix operand order checked by reading, not run on hardware
//
#include <hip/hip_runtime.h>
#include <hip/hip_bf16.h>

typedef __attribute__((ext_vector_type(16))) _Float16 v16h;
typedef __attribute__((ext_vector_type(8)))  float    v8f;

#define THREADS 256
#define BLOCKS  64
#define WAVES_PER_BLOCK (THREADS / 32)
#define TOTAL_WAVES (BLOCKS * WAVES_PER_BLOCK)

// Pass 1: each wave reduces chunks of 512 contiguous elements via
// D = A(|sr-hr| as f16, 16x32) x B(ones) + C, accumulating in the f32 C tile.
__global__ __launch_bounds__(THREADS) void l1_partial_wmma(
    const float* __restrict__ sr, const float* __restrict__ hr,
    int n, float* __restrict__ partials)
{
    const int lane    = threadIdx.x & 31;
    const int gwave   = blockIdx.x * (blockDim.x >> 5) + (threadIdx.x >> 5);
    const int nwaves  = gridDim.x * (blockDim.x >> 5);
    const int nchunks = n >> 9;            // 512 floats per chunk

    v16h ones;
#pragma unroll
    for (int i = 0; i < 16; ++i) ones[i] = (_Float16)1.0f;

    v8f acc = {};

    // Union view of the A-matrix so we can clear f16 sign bits with
    // integer ANDs (avoids llvm.fabs.f32, which crashes gfx1250 ISel
    // when combined with packed-f32 vectorization).
    union AMat {
        v16h h;
        unsigned int u[8];
    };

    for (int chunk = gwave; chunk < nchunks; chunk += nwaves) {
        const float4* a4 = (const float4*)(sr + ((long)chunk << 9));
        const float4* b4 = (const float4*)(hr + ((long)chunk << 9));

        // Prefetch the next chunk this wave will touch (global_prefetch_b8).
        if (chunk + nwaves < nchunks) {
            __builtin_prefetch(sr + ((long)(chunk + nwaves) << 9), 0, 3);
            __builtin_prefetch(hr + ((long)(chunk + nwaves) << 9), 0, 3);
        }

        AMat am;
#pragma unroll
        for (int j = 0; j < 4; ++j) {
            float4 s = a4[lane + 32 * j];   // coalesced: 512B per wave per j
            float4 h = b4[lane + 32 * j];
            // Signed diffs; RTE f32->f16 conversion is sign-symmetric, so
            // clearing the f16 sign bit afterwards equals |s-h| in f16.
            am.h[4 * j + 0] = (_Float16)(s.x - h.x);
            am.h[4 * j + 1] = (_Float16)(s.y - h.y);
            am.h[4 * j + 2] = (_Float16)(s.z - h.z);
            am.h[4 * j + 3] = (_Float16)(s.w - h.w);
        }
#pragma unroll
        for (int w = 0; w < 8; ++w) am.u[w] &= 0x7FFF7FFFu;  // abs of packed f16

        // EXEC is all-ones here (wave-uniform control flow), as WMMA requires.
        acc = __builtin_amdgcn_wmma_f32_16x16x32_f16(
            /*neg_a=*/false, am.h, /*neg_b=*/false, ones,
            /*c_mod=*/(short)0, acc, /*reuse_a=*/false, /*reuse_b=*/false);
    }

    // Every D column replicates the 16 row sums -> total counted 16x.
    float p = acc[0] + acc[1] + acc[2] + acc[3] +
              acc[4] + acc[5] + acc[6] + acc[7];
    p *= (1.0f / 16.0f);

    // Scalar tail for n not a multiple of 512 (none for this problem size).
    const int tail0 = nchunks << 9;
    for (int i = tail0 + (gwave * 32 + lane); i < n; i += nwaves * 32) {
        float d = sr[i] - hr[i];
        p += fmaxf(d, -d);                 // abs via max(d,-d), no llvm.fabs
    }

    // wave32 butterfly reduction
#pragma unroll
    for (int off = 16; off >= 1; off >>= 1)
        p += __shfl_xor(p, off, 32);

    if (lane == 0) partials[gwave] = p;
}

// Pass 2: deterministic single-block reduction of the wave partials.
__global__ __launch_bounds__(THREADS) void l1_finalize(
    const float* __restrict__ partials, int nparts,
    float scale, float* __restrict__ out)
{
    __shared__ float sdata[THREADS];
    float s = 0.0f;
    for (int i = threadIdx.x; i < nparts; i += blockDim.x) s += partials[i];
    sdata[threadIdx.x] = s;
    __syncthreads();
#pragma unroll
    for (int stride = THREADS / 2; stride > 0; stride >>= 1) {
        if ((int)threadIdx.x < stride) sdata[threadIdx.x] += sdata[threadIdx.x + stride];
        __syncthreads();
    }
    if (threadIdx.x == 0) out[0] = sdata[0] * scale;
}

extern "C" void kernel_launch(void* const* d_in, const int* in_sizes, int n_in,
                              void* d_out, int out_size, void* d_ws, size_t ws_size,
                              hipStream_t stream)
{
    // Inputs (setup_inputs order): sr_img f32, hr_img f32, patch_cord i32,
    // h_idx i32, w_idx i32. The index maps cancel analytically (bijective
    // scatter applied identically to both images), so only sr/hr are read.
    const float* sr = (const float*)d_in[0];
    const float* hr = (const float*)d_in[1];
    const int n = in_sizes[0];             // b*c*P*P elements

    float* partials = (float*)d_ws;        // TOTAL_WAVES floats (2 KB)

    l1_partial_wmma<<<BLOCKS, THREADS, 0, stream>>>(sr, hr, n, partials);

    // mean over canvases: sum / (b*c*1024*1024) = sum / (16*n)
    const float scale = 1.0f / (16.0f * (float)n);
    l1_finalize<<<1, THREADS, 0, stream>>>(partials, TOTAL_WAVES, scale,
                                           (float*)d_out);
}
